// GCNLSTMImputer_1185410973710
// MI455X (gfx1250) — compile-verified, matrix operand order checked by reading, not run on hardware
//
#include <hip/hip_runtime.h>
#include <hip/hip_bf16.h>

typedef __attribute__((ext_vector_type(16))) __bf16 v16bf;
typedef __attribute__((ext_vector_type(8)))  float  v8f;
typedef __attribute__((ext_vector_type(4)))  unsigned int v4u;
typedef __attribute__((ext_vector_type(8)))  int v8i;
typedef __attribute__((ext_vector_type(4)))  int v4i;

// ---------- bf16 helpers (bit-level) ----------
static __device__ __forceinline__ unsigned short f32_to_bf16(float f) {
    unsigned int u = __float_as_uint(f);
    u += 0x7FFFu + ((u >> 16) & 1u);   // round-to-nearest-even
    return (unsigned short)(u >> 16);
}
static __device__ __forceinline__ float bf16_to_f32(unsigned short h) {
    return __uint_as_float(((unsigned int)h) << 16);
}

// ---------- TDM: 2D bf16 tile (tile_d1 rows x tile_d0 cols) global -> LDS ----------
// Descriptor per CDNA5 ISA ch.8: group0 {count, lds_addr, global_addr, type=2},
// group1 {wg_mask=0, data_size=1(2B), tensor dims, tile dims, dim0 stride}.
static __device__ __forceinline__ void tdm_load_tile_bf16(
    unsigned lds_off, const unsigned short* gptr,
    unsigned tensor_d0, unsigned tensor_d1,
    unsigned long long row_stride_elems,
    unsigned tile_d0, unsigned tile_d1) {
    unsigned long long ga = (unsigned long long)(uintptr_t)gptr;
    v4u g0;
    g0[0] = 1u;                                        // count=1 valid descriptor
    g0[1] = lds_off;                                   // lds_addr (bytes)
    g0[2] = (unsigned)(ga & 0xFFFFFFFFu);              // global_addr[31:0]
    g0[3] = (unsigned)((ga >> 32) & 0x01FFFFFFu) | (2u << 30); // addr[56:32] | type=2
    v8i g1;
    g1[0] = (int)(1u << 16);                           // wg_mask=0, data_size=1 (2B)
    g1[1] = (int)((tensor_d0 & 0xFFFFu) << 16);        // tensor_dim0[15:0] @bit48
    g1[2] = (int)(((tensor_d0 >> 16) & 0xFFFFu) | ((tensor_d1 & 0xFFFFu) << 16));
    g1[3] = (int)(((tensor_d1 >> 16) & 0xFFFFu) | ((tile_d0 & 0xFFFFu) << 16));
    g1[4] = (int)(tile_d1 & 0xFFFFu);                  // tile_dim1 (tile_dim2=0)
    g1[5] = (int)(row_stride_elems & 0xFFFFFFFFull);   // tensor_dim0_stride[31:0]
    g1[6] = (int)((row_stride_elems >> 32) & 0xFFFFull);
    g1[7] = 0;
    v4i gz = {0, 0, 0, 0};
#if __has_include(<hip/amd_detail/amd_gfx1250_TDM.h>)
    v8i gz8 = {0, 0, 0, 0, 0, 0, 0, 0};
    __builtin_amdgcn_tensor_load_to_lds(g0, g1, gz, gz, gz8, 0);
#else
    __builtin_amdgcn_tensor_load_to_lds(g0, g1, gz, gz, 0);
#endif
}

// ---------- conversion kernels ----------
__global__ void k_conv_bf16(const float* __restrict__ src,
                            unsigned short* __restrict__ dst, int n) {
    int i = blockIdx.x * blockDim.x + threadIdx.x;
    if (i < n) dst[i] = f32_to_bf16(src[i]);
}

// src [rows][cols] f32 -> dst [cols][rows] bf16
__global__ void k_transpose_bf16(const float* __restrict__ src,
                                 unsigned short* __restrict__ dst,
                                 int rows, int cols) {
    int i = blockIdx.x * blockDim.x + threadIdx.x;
    if (i < rows * cols) {
        int r = i / cols, c = i % cols;
        dst[(size_t)c * rows + r] = f32_to_bf16(src[i]);
    }
}

__global__ void k_bias_sum(const float* __restrict__ a, const float* __restrict__ b,
                           float* __restrict__ o, int n) {
    int i = blockIdx.x * blockDim.x + threadIdx.x;
    if (i < n) o[i] = a[i] + b[i];
}

// ---------- tiled bf16 WMMA GEMM:  C[M,N] = A[M,K] * Bm[N,K]^T ----------
// 128x128 block tile, K-step 64, double-buffered LDS. B tiles move via the
// Tensor Data Mover (tensor_load_to_lds / TENSORcnt) prefetched one K-step
// ahead; A tiles are prefetched by VMEM into the alternate buffer while WMMAs
// consume the current one.
// MODE 0: store bf16
// MODE 1: relu, store bf16
// MODE 2: +bias, f32, remap row (b*256+t) -> (t*32+b)   (xproj, t-major for scan)
// MODE 3: +bias, f32, remap row (t*32+b)  -> (b*256+t)  (final output [B,T,N])
template <int MODE>
__global__ __launch_bounds__(256) void gemm_wmma(
    const unsigned short* __restrict__ A,
    const unsigned short* __restrict__ Bm,
    void* __restrict__ Cout,
    const float* __restrict__ bias,
    int M, int N, int K) {
    __shared__ alignas(64) unsigned short As[2][128][64];   // 2 x 16 KB
    __shared__ alignas(64) unsigned short Bs[2][128][64];   // 2 x 16 KB

    const int tid  = threadIdx.x;
    const int lane = tid & 31;
    const int w    = tid >> 5;          // 8 waves
    const int tileM = blockIdx.y * 128;
    const int tileN = blockIdx.x * 128;
    const int w_m = (w & 3) * 32;       // 4 wave rows  (32 M each)
    const int w_n = (w >> 2) * 64;      // 2 wave cols  (64 N each)
    const int lrow = lane & 15;
    const int lhk  = (lane >> 4) * 16;  // K half selected by lane group
    const int ldr  = tid >> 1;          // 0..127 : staging row
    const int ldc  = (tid & 1) * 32;    // 0/32   : staging k-chunk (32 elems)

    auto stageA = [&](int k0, int b) {
        const uint4* sa = reinterpret_cast<const uint4*>(
            A + (size_t)(tileM + ldr) * K + k0 + ldc);
        uint4* da = reinterpret_cast<uint4*>(&As[b][ldr][ldc]);
        da[0] = sa[0]; da[1] = sa[1]; da[2] = sa[2]; da[3] = sa[3];
    };
    auto issueB = [&](int k0, int b) {
        tdm_load_tile_bf16((unsigned)(uintptr_t)(&Bs[b][0][0]),
                           Bm + (size_t)tileN * K + k0,
                           (unsigned)K, (unsigned)N,
                           (unsigned long long)K, 64u, 128u);
    };

    v8f zero = {};
    v8f acc[2][4];
#pragma unroll
    for (int mt = 0; mt < 2; ++mt)
#pragma unroll
        for (int nt = 0; nt < 4; ++nt) acc[mt][nt] = zero;

    const int nIter = K >> 6;
    // prologue: fill buffer 0
    if (tid == 0) issueB(0, 0);
    stageA(0, 0);
    if (tid == 0) __builtin_amdgcn_s_wait_tensorcnt(0);
    __syncthreads();

    int buf = 0;
    for (int it = 0; it < nIter; ++it) {
        // prefetch next K-step into the other buffer (async w.r.t. compute)
        if (it + 1 < nIter) {
            const int k0n = (it + 1) << 6;
            if (tid == 0) issueB(k0n, buf ^ 1);
            stageA(k0n, buf ^ 1);
        }
        // compute on current buffer: 2 k-steps x (2x4 tiles) = 16 WMMAs/wave
#pragma unroll
        for (int ks = 0; ks < 64; ks += 32) {
            v16bf af[2], bfr[4];
#pragma unroll
            for (int mt = 0; mt < 2; ++mt)
                af[mt] = *reinterpret_cast<const v16bf*>(
                    &As[buf][w_m + mt * 16 + lrow][ks + lhk]);
#pragma unroll
            for (int nt = 0; nt < 4; ++nt)
                bfr[nt] = *reinterpret_cast<const v16bf*>(
                    &Bs[buf][w_n + nt * 16 + lrow][ks + lhk]);
#pragma unroll
            for (int mt = 0; mt < 2; ++mt)
#pragma unroll
                for (int nt = 0; nt < 4; ++nt)
                    acc[mt][nt] = __builtin_amdgcn_wmma_f32_16x16x32_bf16(
                        false, af[mt], false, bfr[nt], (short)0, acc[mt][nt],
                        false, false);
        }
        // drain the prefetch before swapping buffers
        if (tid == 0) __builtin_amdgcn_s_wait_tensorcnt(0);
        __syncthreads();
        buf ^= 1;
    }

    // epilogue: C/D layout -> VGPR r, lanes 0-15: M=r col=lane; lanes 16-31: M=r+8
#pragma unroll
    for (int mt = 0; mt < 2; ++mt)
#pragma unroll
        for (int nt = 0; nt < 4; ++nt) {
            const int rbase = tileM + w_m + mt * 16 + ((lane >> 4) << 3);
            const int col   = tileN + w_n + nt * 16 + lrow;
#pragma unroll
            for (int r = 0; r < 8; ++r) {
                const int row = rbase + r;
                float v = acc[mt][nt][r];
                if constexpr (MODE == 0) {
                    ((unsigned short*)Cout)[(size_t)row * N + col] = f32_to_bf16(v);
                } else if constexpr (MODE == 1) {
                    v = v > 0.f ? v : 0.f;
                    ((unsigned short*)Cout)[(size_t)row * N + col] = f32_to_bf16(v);
                } else if constexpr (MODE == 2) {
                    v += bias[col];
                    int t = row & 255, b = row >> 8;
                    ((float*)Cout)[((size_t)t * 32 + b) * N + col] = v;
                } else {
                    v += bias[col];
                    int t = row >> 5, b = row & 31;
                    ((float*)Cout)[((size_t)b * 256 + t) * N + col] = v;
                }
            }
        }
}

// ---------- persistent single-workgroup LSTM scan ----------
// whh: [512][128] bf16 ; xproj: [256][32][512] f32 ; hs: [256][32][128] bf16
__global__ __launch_bounds__(256) void lstm_scan(
    const unsigned short* __restrict__ whh,
    const float* __restrict__ xproj,
    unsigned short* __restrict__ hs) {
    __shared__ alignas(64) unsigned short h_sh[32][128];   // 8 KB
    __shared__ alignas(64) unsigned short gates[32][512];  // 32 KB (bf16 pre-act)

    const int tid  = threadIdx.x;
    const int lane = tid & 31;
    const int w    = tid >> 5;
    const int lrow = lane & 15;
    const int lhk  = (lane >> 4) * 16;

    for (int i = tid; i < 32 * 128; i += 256) (&h_sh[0][0])[i] = 0;
    float c[16];
#pragma unroll
    for (int i = 0; i < 16; ++i) c[i] = 0.f;
    const int e0 = tid * 16;
    const int eb = e0 >> 7;      // batch row this thread owns
    const int ej = e0 & 127;     // hidden offset
    __syncthreads();

    const int mt    = w & 1;            // M tile (rows mt*16..)
    const int nbase = (w >> 1) * 128;   // 8 N-tiles per wave

    for (int t = 0; t < 256; ++t) {
        v16bf af[4];
#pragma unroll
        for (int ks = 0; ks < 4; ++ks)
            af[ks] = *reinterpret_cast<const v16bf*>(
                &h_sh[mt * 16 + lrow][ks * 32 + lhk]);
#pragma unroll
        for (int nt = 0; nt < 8; ++nt) {
            const int n0 = nbase + nt * 16;
            v8f acc = {};
#pragma unroll
            for (int ks = 0; ks < 4; ++ks) {
                v16bf bfr = *reinterpret_cast<const v16bf*>(
                    &whh[(size_t)(n0 + lrow) * 128 + ks * 32 + lhk]);
                acc = __builtin_amdgcn_wmma_f32_16x16x32_bf16(
                    false, af[ks], false, bfr, (short)0, acc, false, false);
            }
#pragma unroll
            for (int r = 0; r < 8; ++r) {
                const int b = mt * 16 + ((lane >> 4) << 3) + r;
                const int n = n0 + lrow;
                float v = acc[r] + xproj[((size_t)t * 32 + b) * 512 + n];
                gates[b][n] = f32_to_bf16(v);
            }
        }
        __syncthreads();
        // elementwise gate math: i,f,g,o packed along 4*LH
#pragma unroll
        for (int i = 0; i < 16; ++i) {
            const int j = ej + i;
            float ig = bf16_to_f32(gates[eb][j]);
            float fg = bf16_to_f32(gates[eb][128 + j]);
            float gg = bf16_to_f32(gates[eb][256 + j]);
            float og = bf16_to_f32(gates[eb][384 + j]);
            float is = 1.f / (1.f + __expf(-ig));
            float fs = 1.f / (1.f + __expf(-fg));
            float gt = tanhf(gg);
            float os = 1.f / (1.f + __expf(-og));
            float cn = fs * c[i] + is * gt;
            c[i] = cn;
            unsigned short hb = f32_to_bf16(os * tanhf(cn));
            h_sh[eb][j] = hb;
            hs[((size_t)t * 32 + eb) * 128 + j] = hb;
        }
        __syncthreads();
    }
}

extern "C" void kernel_launch(void* const* d_in, const int* in_sizes, int n_in,
                              void* d_out, int out_size, void* d_ws, size_t ws_size,
                              hipStream_t stream) {
    const float* x     = (const float*)d_in[0];  // [32,256,2048]
    const float* adj   = (const float*)d_in[1];  // [2048,2048]
    const float* gcn_w = (const float*)d_in[2];  // [2048,128]
    const float* w_ih  = (const float*)d_in[3];  // [512,128]
    const float* w_hh  = (const float*)d_in[4];  // [512,128]
    const float* b_ih  = (const float*)d_in[5];  // [512]
    const float* b_hh  = (const float*)d_in[6];  // [512]
    const float* fc_w  = (const float*)d_in[7];  // [2048,128]
    const float* fc_b  = (const float*)d_in[8];  // [2048]
    float* out = (float*)d_out;                  // [32,256,2048]

    char* ws = (char*)d_ws;
    size_t off = 0;
    auto alloc = [&](size_t bytes) -> char* {
        char* p = ws + off;
        off = (off + bytes + 255) & ~(size_t)255;
        return p;
    };
    unsigned short* xb    = (unsigned short*)alloc((size_t)8192 * 2048 * 2); // 32 MB
    unsigned short* adjb  = (unsigned short*)alloc((size_t)2048 * 2048 * 2); //  8 MB
    unsigned short* aggb  = (unsigned short*)alloc((size_t)8192 * 2048 * 2); // 32 MB
    unsigned short* gwT   = (unsigned short*)alloc((size_t)128 * 2048 * 2);
    unsigned short* gcnb  = (unsigned short*)alloc((size_t)8192 * 128 * 2);
    unsigned short* wihb  = (unsigned short*)alloc((size_t)512 * 128 * 2);
    unsigned short* whhb  = (unsigned short*)alloc((size_t)512 * 128 * 2);
    float*          biasb = (float*)alloc(512 * 4);
    float*          xproj = (float*)alloc((size_t)256 * 32 * 512 * 4);       // 16 MB
    unsigned short* hsb   = (unsigned short*)alloc((size_t)8192 * 128 * 2);
    unsigned short* fcwb  = (unsigned short*)alloc((size_t)2048 * 128 * 2);

    const int TPB = 256;
    k_conv_bf16<<<(16777216 + TPB - 1) / TPB, TPB, 0, stream>>>(x, xb, 16777216);
    k_conv_bf16<<<(4194304 + TPB - 1) / TPB, TPB, 0, stream>>>(adj, adjb, 4194304);
    k_conv_bf16<<<(65536 + TPB - 1) / TPB, TPB, 0, stream>>>(w_ih, wihb, 65536);
    k_conv_bf16<<<(65536 + TPB - 1) / TPB, TPB, 0, stream>>>(w_hh, whhb, 65536);
    k_conv_bf16<<<(262144 + TPB - 1) / TPB, TPB, 0, stream>>>(fc_w, fcwb, 262144);
    k_transpose_bf16<<<(262144 + TPB - 1) / TPB, TPB, 0, stream>>>(gcn_w, gwT, 2048, 128);
    k_bias_sum<<<2, TPB, 0, stream>>>(b_ih, b_hh, biasb, 512);

    // agg = x @ adj^T               [8192,2048] = [8192,2048]x[2048,2048]^T
    gemm_wmma<0><<<dim3(16, 64), TPB, 0, stream>>>(xb, adjb, aggb, nullptr,
                                                   8192, 2048, 2048);
    // gcn = relu(agg @ gcn_w)       [8192,128]
    gemm_wmma<1><<<dim3(1, 64), TPB, 0, stream>>>(aggb, gwT, gcnb, nullptr,
                                                  8192, 128, 2048);
    // xproj = gcn @ w_ih^T + (b_ih+b_hh), re-laid as [T,B,4LH]
    gemm_wmma<2><<<dim3(4, 64), TPB, 0, stream>>>(gcnb, wihb, xproj, biasb,
                                                  8192, 512, 128);
    // sequential LSTM scan (single persistent workgroup)
    lstm_scan<<<1, TPB, 0, stream>>>(whhb, xproj, hsb);
    // out = hs @ fc_w^T + fc_b, rows remapped [T,B] -> [B,T]
    gemm_wmma<3><<<dim3(16, 64), TPB, 0, stream>>>(hsb, fcwb, out, fc_b,
                                                   8192, 2048, 128);
}
